// FNPGNNv4_55052890800674
// MI455X (gfx1250) — compile-verified, matrix operand order checked by reading, not run on hardware
//
#include <hip/hip_runtime.h>
#include <hip/hip_bf16.h>
#include <math.h>

// ---------------- problem constants ----------------
#define BB    512
#define NNODE 116
#define HD    128
#define BN    (BB * NNODE)      // 59392, divisible by 16
#define HEADS 4
#define DH    32
#define INDIM 123
#define FLATD 1280

typedef __attribute__((ext_vector_type(16))) _Float16 v16h;
typedef __attribute__((ext_vector_type(8)))  float    v8f;

#define DEVINL __device__ __forceinline__

// ---- optional CDNA5 async global->LDS path (guarded: falls back cleanly) ----
#if defined(__has_builtin)
#if __has_builtin(__builtin_amdgcn_global_load_async_to_lds_b128) && \
    __has_builtin(__builtin_amdgcn_s_wait_asynccnt)
#define HAVE_ASYNC_LDS 1
#endif
#endif

#if defined(HAVE_ASYNC_LDS)
typedef int asv4i __attribute__((vector_size(16)));
typedef __attribute__((address_space(1))) asv4i* gp4i;
typedef __attribute__((address_space(3))) asv4i* lp4i;
DEVINL void async_cp16(const _Float16* g, _Float16* l) {
  __builtin_amdgcn_global_load_async_to_lds_b128((gp4i)(void*)(size_t)(const void*)g,
                                                 (lp4i)(void*)l, 0, 0);
}
DEVINL void async_wait0() { __builtin_amdgcn_s_wait_asynccnt(0); }
#endif

DEVINL v8f wmma16x16x32(v16h a, v16h b, v8f c) {
  // D = A(16x32,f16) * B(32x16,f16) + C(16x16,f32)
  return __builtin_amdgcn_wmma_f32_16x16x32_f16(false, a, false, b, (short)0, c,
                                                false, false);
}

// Fragment loader: 16 halves for one K-chunk of a row/col
// (elements 0..7 = K kb..kb+7, 8..15 = K kb+16..kb+23 per ISA 7.12.2)
DEVINL v16h load_frag(const _Float16* base, int kb) {
  v16h f;
  const _Float16* p = base + kb;
#pragma unroll
  for (int t = 0; t < 8; ++t) { f[t] = p[t]; f[8 + t] = p[16 + t]; }
  return f;
}

DEVINL float elu_f(float x) { return x > 0.f ? x : (expf(x) - 1.f); }

// ---------------- x f32 [BN,123] -> f16 [BN,128] zero-padded
__global__ void k_prep_x(const float* __restrict__ x, _Float16* __restrict__ xh) {
  size_t idx = (size_t)blockIdx.x * blockDim.x + threadIdx.x;
  if (idx >= (size_t)BN * 128) return;
  int k = (int)(idx & 127);
  size_t row = idx >> 7;
  xh[idx] = (k < INDIM) ? (_Float16)x[row * INDIM + k] : (_Float16)0.f;
}

// ---------------- weight prep: f32 [K,N] -> f16 [N][Kpad] (transposed, zero-pad K)
__global__ void k_prep_wt(const float* __restrict__ W, _Float16* __restrict__ Wt,
                          int Krows, int Ncols, int Kpad) {
  int idx = blockIdx.x * blockDim.x + threadIdx.x;
  if (idx >= Ncols * Kpad) return;
  int col = idx / Kpad, k = idx - col * Kpad;
  Wt[idx] = (k < Krows) ? (_Float16)W[(size_t)k * Ncols + col] : (_Float16)0.f;
}

// ---------------- encoder: h = BN(elu(x @ W_enc + b)) ; writes f32 + f16
__global__ void __launch_bounds__(256) k_encoder(
    const _Float16* __restrict__ xh, const _Float16* __restrict__ WT,
    const float* __restrict__ be, const float* __restrict__ gg,
    const float* __restrict__ bb, const float* __restrict__ bm,
    const float* __restrict__ bvv, float* __restrict__ A,
    _Float16* __restrict__ Ah) {
  __shared__ __align__(16) _Float16 hs[16 * 128];
  int tid = threadIdx.x;
  int r0 = blockIdx.x * 16;
#if defined(HAVE_ASYNC_LDS)
  async_cp16(xh + (size_t)r0 * 128 + tid * 8, hs + tid * 8);
  async_wait0();
#else
  for (int i = tid; i < 16 * 128; i += 256)
    hs[i] = xh[(size_t)(r0 + (i >> 7)) * 128 + (i & 127)];
#endif
  __syncthreads();
  int wave = tid >> 5, lane = tid & 31;
  int n0 = wave * 16, rl = lane & 15, g8 = (lane >> 4) << 3;
  int col = n0 + rl;
  const _Float16* wcol = WT + (size_t)col * 128;
  const _Float16* arow = hs + rl * 128;
  __builtin_prefetch(wcol, 0, 0);
  v8f acc = {};
#pragma unroll
  for (int c = 0; c < 4; ++c) {
    int kb = c * 32 + g8;
    acc = wmma16x16x32(load_frag(arow, kb), load_frag(wcol, kb), acc);
  }
  float bias = be[col], g = gg[col], b2 = bb[col], mm = bm[col];
  float rs = rsqrtf(bvv[col] + 1e-5f);
#pragma unroll
  for (int v = 0; v < 8; ++v) {
    int row = r0 + v + g8;
    float val = elu_f(acc[v] + bias);
    val = (val - mm) * rs * g + b2;
    size_t o = (size_t)row * 128 + col;
    A[o] = val;
    Ah[o] = (_Float16)val;
  }
}

// ---------------- Wh = h @ W  (128x128), f16 out
__global__ void __launch_bounds__(256) k_wh(const _Float16* __restrict__ Hh,
                                            const _Float16* __restrict__ WT,
                                            _Float16* __restrict__ Wbh) {
  __shared__ __align__(16) _Float16 hs[16 * 128];
  int tid = threadIdx.x;
  int r0 = blockIdx.x * 16;
#if defined(HAVE_ASYNC_LDS)
  async_cp16(Hh + (size_t)r0 * 128 + tid * 8, hs + tid * 8);
  async_wait0();
#else
  for (int i = tid; i < 16 * 128; i += 256)
    hs[i] = Hh[(size_t)(r0 + (i >> 7)) * 128 + (i & 127)];
#endif
  __syncthreads();
  int wave = tid >> 5, lane = tid & 31;
  int n0 = wave * 16, rl = lane & 15, g8 = (lane >> 4) << 3;
  int col = n0 + rl;
  const _Float16* wcol = WT + (size_t)col * 128;
  const _Float16* arow = hs + rl * 128;
  __builtin_prefetch(wcol, 0, 0);
  v8f acc = {};
#pragma unroll
  for (int c = 0; c < 4; ++c) {
    int kb = c * 32 + g8;
    acc = wmma16x16x32(load_frag(arow, kb), load_frag(wcol, kb), acc);
  }
#pragma unroll
  for (int v = 0; v < 8; ++v)
    Wbh[(size_t)(r0 + v + g8) * 128 + col] = (_Float16)acc[v];
}

// ---------------- ei/ej = per-head dots of Wh with attention vector a
__global__ void k_eiej(const _Float16* __restrict__ Wbh, const float* __restrict__ a,
                       float* __restrict__ ei, float* __restrict__ ej) {
  int idx = blockIdx.x * blockDim.x + threadIdx.x;
  if (idx >= BN * HEADS) return;
  int row = idx >> 2, h = idx & 3;
  const _Float16* p = Wbh + (size_t)row * 128 + h * DH;
  const float* ai = a + h * (2 * DH);
  float si = 0.f, sj = 0.f;
#pragma unroll
  for (int d = 0; d < DH; ++d) {
    float w = (float)p[d];
    si += w * ai[d];
    sj += w * ai[DH + d];
  }
  ei[idx] = si;
  ej[idx] = sj;
}

// ---------------- attention: out[b,i,h,:] = softmax_j(mask(leaky(ei+ej))) @ Wh[b,:,h,:]
// Wh head slice staged TRANSPOSED in LDS ([d][j]) so B fragments are two
// contiguous 16-byte ds loads per chunk.
__global__ void __launch_bounds__(256) k_attn(
    const float* __restrict__ ei, const float* __restrict__ ej,
    const float* __restrict__ adj, const _Float16* __restrict__ Wbh,
    float* __restrict__ At) {
  int b = blockIdx.x, h = blockIdx.y;
  __shared__ __align__(16) _Float16 WhT[DH * 128];  // [d][j], j padded to 128
  __shared__ float ejS[128];
  int tid = threadIdx.x;
  for (int i = tid; i < DH * 128; i += 256) {
    int d = i >> 7, j = i & 127;
    WhT[i] = (j < NNODE) ? Wbh[((size_t)b * NNODE + j) * 128 + h * DH + d]
                         : (_Float16)0.f;
  }
  for (int j = tid; j < 128; j += 256)
    ejS[j] = (j < NNODE) ? ej[((size_t)b * NNODE + j) * HEADS + h] : 0.f;
  __syncthreads();

  int wave = tid >> 5, lane = tid & 31;
  int rl = lane & 15, g8 = (lane >> 4) << 3;
  int i = wave * 16 + rl;
  bool vi = (i < NNODE);
  float eival = vi ? ei[((size_t)b * NNODE + i) * HEADS + h] : 0.f;
  const float* adjrow = adj + ((size_t)b * NNODE + (vi ? i : 0)) * NNODE;

  // pass 1: row max (lane pair (lane, lane^16) covers all 32 j's of each chunk)
  float mrow = -3.0e38f;
  for (int c = 0; c < 4; ++c) {
    int kb = c * 32 + g8;
#pragma unroll
    for (int t = 0; t < 16; ++t) {
      int j = kb + ((t < 8) ? t : 8 + t);
      float s = -3.0e38f;
      if (vi && j < NNODE && adjrow[j] != 0.f) {
        s = eival + ejS[j];
        s = s > 0.f ? s : 0.2f * s;
      }
      mrow = fmaxf(mrow, s);
    }
  }
  mrow = fmaxf(mrow, __shfl_xor(mrow, 16));

  // pass 2: row sum of exp
  float sum = 0.f;
  for (int c = 0; c < 4; ++c) {
    int kb = c * 32 + g8;
#pragma unroll
    for (int t = 0; t < 16; ++t) {
      int j = kb + ((t < 8) ? t : 8 + t);
      if (vi && j < NNODE && adjrow[j] != 0.f) {
        float s = eival + ejS[j];
        s = s > 0.f ? s : 0.2f * s;
        sum += expf(s - mrow);
      }
    }
  }
  sum += __shfl_xor(sum, 16);
  float inv = (sum > 0.f) ? 1.f / sum : 0.f;

  // pass 3: build P fragments (A) and multiply against Wh (B) via WMMA
  const _Float16* bcol0 = WhT + (size_t)rl * 128;
  const _Float16* bcol1 = WhT + (size_t)(16 + rl) * 128;
  v8f acc0 = {}, acc1 = {};
  for (int c = 0; c < 4; ++c) {
    int kb = c * 32 + g8;
    v16h af;
#pragma unroll
    for (int t = 0; t < 16; ++t) {
      int j = kb + ((t < 8) ? t : 8 + t);
      float p = 0.f;
      if (vi && j < NNODE && adjrow[j] != 0.f) {
        float s = eival + ejS[j];
        s = s > 0.f ? s : 0.2f * s;
        p = expf(s - mrow) * inv;
      }
      af[t] = (_Float16)p;
    }
    acc0 = wmma16x16x32(af, load_frag(bcol0, kb), acc0);
    acc1 = wmma16x16x32(af, load_frag(bcol1, kb), acc1);
  }
#pragma unroll
  for (int v = 0; v < 8; ++v) {
    int ir = wave * 16 + v + g8;
    if (ir < NNODE) {
      size_t o = ((size_t)b * NNODE + ir) * 128 + h * DH;
      At[o + rl] = acc0[v];
      At[o + 16 + rl] = acc1[v];
    }
  }
}

// ---------------- post: h = elu(BN(attn_out)) + Wh [+ h_prev]
__global__ void k_post(const float* __restrict__ At, const _Float16* __restrict__ Wbh,
                       const float* __restrict__ g, const float* __restrict__ bt,
                       const float* __restrict__ m, const float* __restrict__ v,
                       float* __restrict__ A, int addPrev) {
  size_t idx = (size_t)blockIdx.x * blockDim.x + threadIdx.x;
  if (idx >= (size_t)BN * 128) return;
  int d = (int)(idx & 127);
  float val = At[idx];
  val = (val - m[d]) * rsqrtf(v[d] + 1e-5f) * g[d] + bt[d];
  float r = elu_f(val) + (float)Wbh[idx];
  if (addPrev) r += A[idx];
  A[idx] = r;
}

// ---------------- per-graph mean over nodes
__global__ void k_mean(const float* __restrict__ A, float* __restrict__ out) {
  int b = blockIdx.x, d = threadIdx.x;
  float s = 0.f;
  for (int n = 0; n < NNODE; ++n) s += A[((size_t)b * NNODE + n) * 128 + d];
  out[b * 128 + d] = s * (1.f / (float)NNODE);
}

// ---------------- vn = elu(concat(x1(+addp), x2) @ Wv + bv)
__global__ void k_vnmix(const float* __restrict__ x1, const float* __restrict__ addp,
                        const float* __restrict__ x2, const float* __restrict__ Wv,
                        const float* __restrict__ bv, float* __restrict__ out) {
  int b = blockIdx.x, d = threadIdx.x;
  float acc = bv[d];
  for (int k = 0; k < 128; ++k) {
    float a = x1[b * 128 + k] + (addp ? addp[k] : 0.f);
    acc += a * Wv[(size_t)k * 128 + d];
  }
  for (int k = 0; k < 128; ++k)
    acc += x2[b * 128 + k] * Wv[(size_t)(128 + k) * 128 + d];
  out[b * 128 + d] = elu_f(acc);
}

// ---------------- h += 0.1*vn (broadcast over nodes), refresh f16 copy
__global__ void k_addvn(float* __restrict__ A, _Float16* __restrict__ Ah,
                        const float* __restrict__ vn) {
  size_t idx = (size_t)blockIdx.x * blockDim.x + threadIdx.x;
  if (idx >= (size_t)BN * 128) return;
  int d = (int)(idx & 127);
  int b = (int)(idx >> 7) / NNODE;
  float r = A[idx] + 0.1f * vn[b * 128 + d];
  A[idx] = r;
  Ah[idx] = (_Float16)r;
}

// ---------------- pooled[b,k,d] = sum_n h[b,n,d] * pm[n,k]
__global__ void k_pool(const float* __restrict__ A, const float* __restrict__ pm,
                       float* __restrict__ pooled) {
  int b = blockIdx.x, k = blockIdx.y, d = threadIdx.x;
  float s = 0.f;
  for (int n = 0; n < NNODE; ++n)
    s += A[((size_t)b * NNODE + n) * 128 + d] * pm[n * 9 + k];
  pooled[((size_t)b * 9 + k) * 128 + d] = s;
}

// ---------------- pool attention + softmax + flat concat + LayerNorm (per graph)
__global__ void __launch_bounds__(256) k_fuse(
    const float* __restrict__ pooled, const float* __restrict__ Wa1,
    const float* __restrict__ ba1, const float* __restrict__ Wa2,
    const float* __restrict__ ba2, const float* __restrict__ vnf,
    const float* __restrict__ lng, const float* __restrict__ lnb,
    float* __restrict__ outFlat, _Float16* __restrict__ flat16) {
  int b = blockIdx.x, tid = threadIdx.x;
  __shared__ float ps[1152];
  __shared__ float us[288];
  __shared__ float wk[9];
  __shared__ float fs[FLATD];
  __shared__ float red[256], red2[256];
  __shared__ float sstat[2];

  for (int i = tid; i < 1152; i += 256) ps[i] = pooled[(size_t)b * 1152 + i];
  __syncthreads();
  for (int i = tid; i < 288; i += 256) {
    int k = i >> 5, j = i & 31;
    float acc = ba1[j];
    for (int d = 0; d < 128; ++d) acc += ps[k * 128 + d] * Wa1[d * 32 + j];
    us[i] = tanhf(acc);
  }
  __syncthreads();
  if (tid < 9) {
    float acc = ba2[0];
    for (int j = 0; j < 32; ++j) acc += us[tid * 32 + j] * Wa2[j];
    wk[tid] = acc;
  }
  __syncthreads();
  if (tid == 0) {
    float mx = wk[0];
    for (int k = 1; k < 9; ++k) mx = fmaxf(mx, wk[k]);
    float s = 0.f;
    for (int k = 0; k < 9; ++k) { wk[k] = expf(wk[k] - mx); s += wk[k]; }
    float invs = 1.f / s;
    for (int k = 0; k < 9; ++k) wk[k] *= invs;
  }
  __syncthreads();
  for (int i = tid; i < FLATD; i += 256)
    fs[i] = (i < 1152) ? ps[i] * wk[i >> 7] : vnf[b * 128 + (i - 1152)];
  __syncthreads();
  float s1 = 0.f, s2 = 0.f;
  for (int i = tid; i < FLATD; i += 256) { float f = fs[i]; s1 += f; s2 += f * f; }
  red[tid] = s1; red2[tid] = s2;
  __syncthreads();
  for (int st = 128; st > 0; st >>= 1) {
    if (tid < st) { red[tid] += red[tid + st]; red2[tid] += red2[tid + st]; }
    __syncthreads();
  }
  if (tid == 0) {
    float mu = red[0] / (float)FLATD;
    float var = red2[0] / (float)FLATD - mu * mu;
    sstat[0] = mu;
    sstat[1] = rsqrtf(var + 1e-5f);
  }
  __syncthreads();
  float mu = sstat[0], rs = sstat[1];
  for (int i = tid; i < FLATD; i += 256) {
    float v = (fs[i] - mu) * rs * lng[i] + lnb[i];
    outFlat[(size_t)b * FLATD + i] = v;
    flat16[(size_t)b * FLATD + i] = (_Float16)v;
  }
}

// ---------------- classifier layer 1: elu(flat @ Wc1 + bc1), WMMA
__global__ void __launch_bounds__(256) k_cls1(const _Float16* __restrict__ F,
                                              const _Float16* __restrict__ WT,
                                              const float* __restrict__ bc,
                                              float* __restrict__ C1) {
  __shared__ __align__(16) _Float16 hs[16 * FLATD];
  int tid = threadIdx.x;
  int r0 = blockIdx.x * 16;
#if defined(HAVE_ASYNC_LDS)
#pragma unroll
  for (int it = 0; it < 10; ++it)
    async_cp16(F + (size_t)r0 * FLATD + it * 2048 + tid * 8,
               hs + it * 2048 + tid * 8);
  async_wait0();
#else
  for (int i = tid; i < 16 * FLATD; i += 256)
    hs[i] = F[(size_t)(r0 + i / FLATD) * FLATD + (i % FLATD)];
#endif
  __syncthreads();
  int wave = tid >> 5, lane = tid & 31;
  int n0 = ((int)blockIdx.y * 8 + wave) * 16;
  int rl = lane & 15, g8 = (lane >> 4) << 3;
  int col = n0 + rl;
  const _Float16* wcol = WT + (size_t)col * FLATD;
  const _Float16* arow = hs + rl * FLATD;
  __builtin_prefetch(wcol, 0, 0);
  v8f acc = {};
  for (int c = 0; c < FLATD / 32; ++c) {
    int kb = c * 32 + g8;
    acc = wmma16x16x32(load_frag(arow, kb), load_frag(wcol, kb), acc);
  }
  float bias = bc[col];
#pragma unroll
  for (int v = 0; v < 8; ++v)
    C1[(size_t)(r0 + v + g8) * 256 + col] = elu_f(acc[v] + bias);
}

// ---------------- classifier layer 2: logits = c1 @ Wc2 + bc2
__global__ void k_cls2(const float* __restrict__ C1, const float* __restrict__ Wc2,
                       const float* __restrict__ bc2, float* __restrict__ logits) {
  int idx = blockIdx.x * blockDim.x + threadIdx.x;
  if (idx >= BB * 2) return;
  int b = idx >> 1, c = idx & 1;
  float acc = bc2[c];
  for (int j = 0; j < 256; ++j) acc += C1[(size_t)b * 256 + j] * Wc2[j * 2 + c];
  logits[idx] = acc;
}

// =====================================================================
extern "C" void kernel_launch(void* const* d_in, const int* in_sizes, int n_in,
                              void* d_out, int out_size, void* d_ws, size_t ws_size,
                              hipStream_t stream) {
  (void)in_sizes; (void)n_in; (void)out_size; (void)ws_size;
  const float* x      = (const float*)d_in[0];
  const float* adj    = (const float*)d_in[1];
  const float* W_enc  = (const float*)d_in[2];
  const float* b_enc  = (const float*)d_in[3];
  const float* bing   = (const float*)d_in[4];
  const float* binb   = (const float*)d_in[5];
  const float* binm   = (const float*)d_in[6];
  const float* binv   = (const float*)d_in[7];
  const float* vne    = (const float*)d_in[8];
  const float* W1     = (const float*)d_in[9];
  const float* a1     = (const float*)d_in[10];
  const float* g1     = (const float*)d_in[11];
  const float* bt1    = (const float*)d_in[12];
  const float* m1     = (const float*)d_in[13];
  const float* v1     = (const float*)d_in[14];
  const float* W2     = (const float*)d_in[15];
  const float* a2     = (const float*)d_in[16];
  const float* g2     = (const float*)d_in[17];
  const float* bt2    = (const float*)d_in[18];
  const float* m2     = (const float*)d_in[19];
  const float* v2     = (const float*)d_in[20];
  const float* Wv     = (const float*)d_in[21];
  const float* bv     = (const float*)d_in[22];
  const float* pm     = (const float*)d_in[23];
  const float* Wa1    = (const float*)d_in[24];
  const float* ba1    = (const float*)d_in[25];
  const float* Wa2    = (const float*)d_in[26];
  const float* ba2    = (const float*)d_in[27];
  const float* lng    = (const float*)d_in[28];
  const float* lnb    = (const float*)d_in[29];
  const float* Wc1    = (const float*)d_in[30];
  const float* bc1    = (const float*)d_in[31];
  const float* Wc2    = (const float*)d_in[32];
  const float* bc2    = (const float*)d_in[33];

  float* logits  = (float*)d_out;           // [512,2]
  float* outFlat = (float*)d_out + BB * 2;  // [512,1280]

  char* base = (char*)d_ws;
  size_t off = 0;
  auto alloc = [&](size_t bytes) -> void* {
    void* p = base + off;
    off += (bytes + 255) & ~(size_t)255;
    return p;
  };
  float*    A      = (float*)   alloc((size_t)BN * 128 * 4);
  _Float16* Ah     = (_Float16*)alloc((size_t)BN * 128 * 2);
  _Float16* xh     = (_Float16*)alloc((size_t)BN * 128 * 2);
  _Float16* Wbh    = (_Float16*)alloc((size_t)BN * 128 * 2);
  float*    At     = (float*)   alloc((size_t)BN * 128 * 4);
  float*    ei     = (float*)   alloc((size_t)BN * HEADS * 4);
  float*    ej     = (float*)   alloc((size_t)BN * HEADS * 4);
  float*    mean0  = (float*)   alloc((size_t)BB * 128 * 4);
  float*    mean1  = (float*)   alloc((size_t)BB * 128 * 4);
  float*    meanF  = (float*)   alloc((size_t)BB * 128 * 4);
  float*    vnN    = (float*)   alloc((size_t)BB * 128 * 4);
  float*    vnF    = (float*)   alloc((size_t)BB * 128 * 4);
  float*    pooled = (float*)   alloc((size_t)BB * 9 * 128 * 4);
  _Float16* flat16 = (_Float16*)alloc((size_t)BB * FLATD * 2);
  float*    C1     = (float*)   alloc((size_t)BB * 256 * 4);
  _Float16* WencT  = (_Float16*)alloc((size_t)128 * 128 * 2);
  _Float16* W1T    = (_Float16*)alloc((size_t)128 * 128 * 2);
  _Float16* W2T    = (_Float16*)alloc((size_t)128 * 128 * 2);
  _Float16* Wc1T   = (_Float16*)alloc((size_t)256 * FLATD * 2);

  const int MT = BN / 16;        // 3712 row tiles
  const int EW = BN * 128 / 256; // 29696 elementwise blocks

  // input / weight prep (f16, transposed, K-padded)
  k_prep_x<<<EW, 256, 0, stream>>>(x, xh);
  k_prep_wt<<<(128 * 128 + 255) / 256, 256, 0, stream>>>(W_enc, WencT, INDIM, 128, 128);
  k_prep_wt<<<(128 * 128 + 255) / 256, 256, 0, stream>>>(W1, W1T, 128, 128, 128);
  k_prep_wt<<<(128 * 128 + 255) / 256, 256, 0, stream>>>(W2, W2T, 128, 128, 128);
  k_prep_wt<<<(256 * FLATD + 255) / 256, 256, 0, stream>>>(Wc1, Wc1T, FLATD, 256, FLATD);

  // encoder + BN
  k_encoder<<<MT, 256, 0, stream>>>(xh, WencT, b_enc, bing, binb, binm, binv, A, Ah);
  k_mean<<<BB, 128, 0, stream>>>(A, mean0);

  // --- GAT layer 1 ---
  k_wh<<<MT, 256, 0, stream>>>(Ah, W1T, Wbh);
  k_eiej<<<(BN * HEADS + 255) / 256, 256, 0, stream>>>(Wbh, a1, ei, ej);
  k_attn<<<dim3(BB, HEADS), 256, 0, stream>>>(ei, ej, adj, Wbh, At);
  k_post<<<EW, 256, 0, stream>>>(At, Wbh, g1, bt1, m1, v1, A, 0);
  k_mean<<<BB, 128, 0, stream>>>(A, mean1);
  k_vnmix<<<BB, 128, 0, stream>>>(mean0, vne, mean1, Wv, bv, vnN);
  k_addvn<<<EW, 256, 0, stream>>>(A, Ah, vnN);

  // --- GAT layer 2 ---
  k_wh<<<MT, 256, 0, stream>>>(Ah, W2T, Wbh);
  k_eiej<<<(BN * HEADS + 255) / 256, 256, 0, stream>>>(Wbh, a2, ei, ej);
  k_attn<<<dim3(BB, HEADS), 256, 0, stream>>>(ei, ej, adj, Wbh, At);
  k_post<<<EW, 256, 0, stream>>>(At, Wbh, g2, bt2, m2, v2, A, 1);
  k_mean<<<BB, 128, 0, stream>>>(A, meanF);
  k_vnmix<<<BB, 128, 0, stream>>>(vnN, nullptr, meanF, Wv, bv, vnF);

  // pooling + attention + LayerNorm + classifier
  k_pool<<<dim3(BB, 9), 128, 0, stream>>>(A, pm, pooled);
  k_fuse<<<BB, 256, 0, stream>>>(pooled, Wa1, ba1, Wa2, ba2, vnF, lng, lnb,
                                 outFlat, flat16);
  k_cls1<<<dim3(BB / 16, 2), 256, 0, stream>>>(flat16, Wc1T, bc1, C1);
  k_cls2<<<(BB * 2 + 255) / 256, 256, 0, stream>>>(C1, Wc2, bc2, logits);
}
